// MessageBlock_18923625906599
// MI455X (gfx1250) — matrix operand order repaired and checked by reading, not compiled, not run on hardware
//
#include <hip/hip_runtime.h>
#include <hip/hip_bf16.h>
#include <math.h>

typedef __attribute__((ext_vector_type(16))) _Float16 v16h;
typedef __attribute__((ext_vector_type(8)))  float    v8f;

#define N_NODES 50000
#define CH      128
#define C3      384
#define NEDGE   400000
#define RCUT    5.0f
#define PI_F    3.14159265358979f

// Packed-fragment tile counts (each tile = 32 lanes x 16 f16 = 512 f16 = 1 KB)
#define W1_KT 4
#define W1_NT 8
#define W2_KT 4
#define W2_NT 24
#define WR_NT 24
#define W1_TILES (W1_KT * W1_NT)   // 32
#define W2_TILES (W2_KT * W2_NT)   // 96
#define WR_TILES (WR_NT)           // 24

static __device__ __forceinline__ v8f wmma_f16(const v16h& a, const v16h& b, const v8f& c) {
    return __builtin_amdgcn_wmma_f32_16x16x32_f16(false, a, false, b, (short)0, c, false, false);
}

// One contiguous 32-byte per-lane load (2x global_load_b128), no conversion.
static __device__ __forceinline__ v16h load_frag(const _Float16* __restrict__ base,
                                                 int tile, int lane) {
    return *(const v16h*)(base + (size_t)tile * 512 + lane * 16);
}

// Fast single-instruction reciprocal (v_rcp_f32); ~1 ulp, far below the f16
// quantization already applied to WMMA operands.
static __device__ __forceinline__ float fast_rcp(float x) {
    return __builtin_amdgcn_rcpf(x);
}

// ---------------------------------------------------------------------------
// Kernel 0: pre-swizzle W1 / W2 / Wr into f16 WMMA B-fragment order.
// B 32x16 f16 layout: lane (half=lane>>4, lm=lane&15), element e holds
// W[kb + (e&7) + 8*half + 16*(e>>3)][nt*16 + lm].  Stored per tile as
// [lane][16 values] contiguous -> fragment load is 32 coalesced bytes/lane.
// ---------------------------------------------------------------------------
__launch_bounds__(256)
__global__ void pack_weights_kernel(const float* __restrict__ W1,
                                    const float* __restrict__ W2,
                                    const float* __restrict__ Wr,
                                    _Float16* __restrict__ w1h,
                                    _Float16* __restrict__ w2h,
                                    _Float16* __restrict__ wrh)
{
    int idx = blockIdx.x * blockDim.x + threadIdx.x;     // one f16 value each
    const int total = (W1_TILES + W2_TILES + WR_TILES) * 512;
    if (idx >= total) return;
    int tile = idx >> 9;
    int pos  = idx & 511;
    int lane = pos >> 4;
    int e    = pos & 15;
    int j    = e >> 1;
    int q    = e & 1;
    int half = lane >> 4;
    int lm   = lane & 15;

    const float* W; _Float16* out; int nT, ncols, kreal, tloc;
    if (tile < W1_TILES) {
        W = W1; out = w1h; nT = W1_NT; ncols = CH;  kreal = CH; tloc = tile;
    } else if (tile < W1_TILES + W2_TILES) {
        W = W2; out = w2h; nT = W2_NT; ncols = C3;  kreal = CH; tloc = tile - W1_TILES;
    } else {
        W = Wr; out = wrh; nT = WR_NT; ncols = C3;  kreal = 20; tloc = tile - W1_TILES - W2_TILES;
    }
    int kb = tloc / nT, nt = tloc % nT;
    int k  = kb * 32 + (e & 7) + 8 * half + 16 * (e >> 3);
    int n  = nt * 16 + lm;
    float val = (k < kreal) ? W[(size_t)k * ncols + n] : 0.0f;
    out[(size_t)tloc * 512 + lane * 16 + e] = (_Float16)val;
}

// ---------------------------------------------------------------------------
// Kernel 1: s_pass = silu(s @ W1 + b1) @ W2 + b2      (N x 384, fp32 into ws)
// 64 node rows per block, 8 waves, WMMA 16x16x32 f16 / fp32 accumulate.
// ---------------------------------------------------------------------------
__launch_bounds__(256)
__global__ void node_mlp_kernel(const float* __restrict__ s,
                                const _Float16* __restrict__ w1h,
                                const _Float16* __restrict__ w2h,
                                const float* __restrict__ b1,
                                const float* __restrict__ b2,
                                float* __restrict__ spass)
{
    __shared__ _Float16 At[64][CH + 8];   // s tile (f16)
    __shared__ _Float16 Ht[64][CH + 8];   // hidden tile (f16)

    const int tid  = threadIdx.x;
    const int lane = tid & 31;
    const int wave = tid >> 5;
    const int half = lane >> 4;
    const int lm   = lane & 15;
    const int rowBase = blockIdx.x * 64;

    // Stage s tile -> LDS (f16), zero-pad tail rows
    for (int i = tid; i < 64 * CH; i += 256) {
        int r = i >> 7;
        int c = i & (CH - 1);
        int node = rowBase + r;
        float val = (node < N_NODES) ? s[(size_t)node * CH + c] : 0.0f;
        At[r][c] = (_Float16)val;
    }
    __syncthreads();

    // ---- GEMM1: H = silu(A @ W1 + b1); 4x8 tiles of 16x16, 4 per wave ----
    for (int t = wave * 4; t < wave * 4 + 4; ++t) {
        const int mt = t >> 3;
        const int nt = t & 7;
        v8f acc = {};
        #pragma unroll
        for (int kb = 0; kb < W1_KT; ++kb) {
            v16h a;
            #pragma unroll
            for (int j = 0; j < 8; ++j) {           // contiguous per half-lane -> ds_load_b128
                int k = kb * 32 + ((j & 3) << 1) + ((j >> 2) << 4) + (half << 3);
                a[2 * j]     = At[mt * 16 + lm][k];
                a[2 * j + 1] = At[mt * 16 + lm][k + 1];
            }
            v16h b = load_frag(w1h, kb * W1_NT + nt, lane);
            acc = wmma_f16(a, b, acc);
        }
        const int n = nt * 16 + lm;
        const float bias = b1[n];
        #pragma unroll
        for (int r = 0; r < 8; ++r) {               // C/D: lane=N, vgpr r -> M=r+8*half
            int m = mt * 16 + r + half * 8;
            float h = acc[r] + bias;
            float sg = fast_rcp(1.0f + __expf(-h)); // v_rcp_f32, not IEEE div chain
            Ht[m][n] = (_Float16)(h * sg);
        }
    }
    __syncthreads();

    // ---- GEMM2: out = H @ W2 + b2; 4x24 tiles, 12 per wave ----
    for (int t = wave * 12; t < wave * 12 + 12; ++t) {
        const int mt = t / 24;
        const int nt = t % 24;
        v8f acc = {};
        #pragma unroll
        for (int kb = 0; kb < W2_KT; ++kb) {
            v16h a;
            #pragma unroll
            for (int j = 0; j < 8; ++j) {
                int k = kb * 32 + ((j & 3) << 1) + ((j >> 2) << 4) + (half << 3);
                a[2 * j]     = Ht[mt * 16 + lm][k];
                a[2 * j + 1] = Ht[mt * 16 + lm][k + 1];
            }
            v16h b = load_frag(w2h, kb * W2_NT + nt, lane);
            acc = wmma_f16(a, b, acc);
        }
        const int n = nt * 16 + lm;
        const float bias = b2[n];
        #pragma unroll
        for (int r = 0; r < 8; ++r) {
            int m = mt * 16 + r + half * 8;
            int node = rowBase + m;
            if (node < N_NODES)
                spass[(size_t)node * C3 + n] = acc[r] + bias;
        }
    }
}

// ---------------------------------------------------------------------------
// Kernel 2: per-edge rbf GEMM (WMMA, M=16 edges, K=20->32) fused with
// gather of s_pass[src]/v[src] and atomic scatter into s_out/v_out.
// 128 edges per block (E = 3125 * 128 exactly), 8 waves x 16 edges.
// ---------------------------------------------------------------------------
__launch_bounds__(256)
__global__ void edge_message_kernel(const float* __restrict__ v,
                                    const long long* __restrict__ edges,
                                    const float* __restrict__ r_ij,
                                    const float* __restrict__ rhat,
                                    const _Float16* __restrict__ wrh,
                                    const float* __restrict__ br,
                                    const float* __restrict__ spass,
                                    float* __restrict__ s_out,
                                    float* __restrict__ v_out)
{
    __shared__ int   srcA[128];
    __shared__ int   dstA[128];
    __shared__ float rA[128];
    __shared__ float fcA[128];
    __shared__ float rhA[128][3];

    const int tid  = threadIdx.x;
    const int lane = tid & 31;
    const int wave = tid >> 5;
    const int half = lane >> 4;
    const int lm   = lane & 15;
    const int eBase = blockIdx.x * 128;

    // Stage per-edge metadata; prefetch the L2 gather lines we will need.
    if (tid < 128) {
        long long e = eBase + tid;
        int dn = (int)edges[2 * e + 0];
        int sn = (int)edges[2 * e + 1];
        dstA[tid] = dn;
        srcA[tid] = sn;
        float r = r_ij[e];
        rA[tid]  = r;
        fcA[tid] = 0.5f * (__cosf((PI_F / RCUT) * r) + 1.0f);
        rhA[tid][0] = rhat[3 * e + 0];
        rhA[tid][1] = rhat[3 * e + 1];
        rhA[tid][2] = rhat[3 * e + 2];
        const float* sp = spass + (size_t)sn * C3;   // 1536 B row
        __builtin_prefetch(sp,       0, 1);
        __builtin_prefetch(sp + 128, 0, 1);
        __builtin_prefetch(sp + 256, 0, 1);
        const float* vp = v + (size_t)sn * C3;
        __builtin_prefetch(vp,       0, 1);
        __builtin_prefetch(vp + 192, 0, 1);
    }
    __syncthreads();

    const int weBase = wave * 16;           // this wave's 16 edges
    const float re   = rA[weBase + lm];     // lane's edge radius (A row = lm)
    const float rinv = fast_rcp(re);        // r in [0.1, 5] -> v_rcp_f32 is fine

    // A fragment: rbf features, 16 edges x 32 K (12 zero pad)
    v16h a;
    #pragma unroll
    for (int j = 0; j < 8; ++j) {
        int k0 = ((j & 3) << 1) + ((j >> 2) << 4) + (half << 3);
        #pragma unroll
        for (int q = 0; q < 2; ++q) {
            int k = k0 + q;
            float val = (k < 20) ? __sinf((PI_F / RCUT) * (float)(k + 1) * re) * rinv : 0.0f;
            a[2 * j + q] = (_Float16)val;
        }
    }

    // 8 column tiles of 16; dv/ds/drep computed together so the two v_out
    // contributions (dv, drep) fuse into one atomic per element.
    for (int ct = 0; ct < 8; ++ct) {
        v16h bv  = load_frag(wrh, ct,      lane);
        v16h bs  = load_frag(wrh, ct + 8,  lane);
        v16h brp = load_frag(wrh, ct + 16, lane);

        v8f av = {}, as_ = {}, ar = {};
        av  = wmma_f16(a, bv,  av);
        as_ = wmma_f16(a, bs,  as_);
        ar  = wmma_f16(a, brp, ar);

        const int c = ct * 16 + lm;
        const float biasV = br[c];
        const float biasS = br[128 + c];
        const float biasR = br[256 + c];
        #pragma unroll
        for (int r = 0; r < 8; ++r) {
            int el = weBase + r + half * 8;     // edge row in this block
            int sN = srcA[el];
            int dN = dstA[el];
            float fc = fcA[el];
            size_t sp = (size_t)sN * C3;
            float pv = (av[r]  + biasV) * fc * spass[sp + c];          // dv
            float ps = (as_[r] + biasS) * fc * spass[sp + 128 + c];    // ds
            float pr = (ar[r]  + biasR) * fc * spass[sp + 256 + c];    // drep

            atomicAdd(&s_out[(size_t)dN * CH + c], ps);
            size_t vb = (size_t)sN * C3;
            size_t ob = (size_t)dN * C3;
            #pragma unroll
            for (int d = 0; d < 3; ++d) {
                float dvv = v[vb + d * CH + c] * pv + rhA[el][d] * pr;
                atomicAdd(&v_out[ob + d * CH + c], dvv);
            }
        }
    }
}

extern "C" void kernel_launch(void* const* d_in, const int* in_sizes, int n_in,
                              void* d_out, int out_size, void* d_ws, size_t ws_size,
                              hipStream_t stream)
{
    const float*     s     = (const float*)d_in[0];
    const float*     v     = (const float*)d_in[1];
    const long long* edges = (const long long*)d_in[2];
    const float*     r_ij  = (const float*)d_in[3];
    const float*     rhat  = (const float*)d_in[4];
    const float*     W1    = (const float*)d_in[5];
    const float*     b1    = (const float*)d_in[6];
    const float*     W2    = (const float*)d_in[7];
    const float*     b2    = (const float*)d_in[8];
    const float*     Wr    = (const float*)d_in[9];
    const float*     br    = (const float*)d_in[10];

    float* s_out = (float*)d_out;                         // N x 128
    float* v_out = (float*)d_out + (size_t)N_NODES * CH;  // N x 3 x 128

    // Workspace layout: spass (N x 384 fp32), then packed f16 weight tiles.
    float*    spass = (float*)d_ws;
    _Float16* w1h   = (_Float16*)((char*)d_ws + (size_t)N_NODES * C3 * sizeof(float));
    _Float16* w2h   = w1h + (size_t)W1_TILES * 512;
    _Float16* wrh   = w2h + (size_t)W2_TILES * 512;

    // Identity terms: s_out = s, v_out = v (graph-capture-safe d2d copies).
    hipMemcpyAsync(s_out, s, (size_t)N_NODES * CH * sizeof(float),
                   hipMemcpyDeviceToDevice, stream);
    hipMemcpyAsync(v_out, v, (size_t)N_NODES * C3 * sizeof(float),
                   hipMemcpyDeviceToDevice, stream);

    dim3 blk(256);
    const int packTotal = (W1_TILES + W2_TILES + WR_TILES) * 512;
    pack_weights_kernel<<<(packTotal + 255) / 256, blk, 0, stream>>>(W1, W2, Wr,
                                                                     w1h, w2h, wrh);
    node_mlp_kernel<<<(N_NODES + 63) / 64, blk, 0, stream>>>(s, w1h, w2h, b1, b2, spass);
    edge_message_kernel<<<NEDGE / 128, blk, 0, stream>>>(v, edges, r_ij, rhat, wrh, br,
                                                         spass, s_out, v_out);
}
